// QuantRNNModel_29970281791588
// MI455X (gfx1250) — matrix operand order repaired and checked by reading, not compiled, not run on hardware
//
#include <hip/hip_runtime.h>
#include <hip/hip_bf16.h>

typedef __attribute__((ext_vector_type(16))) _Float16 v16h;
typedef __attribute__((ext_vector_type(8)))  float    v8f;
typedef int v4i_b __attribute__((vector_size(4 * sizeof(int))));  // matches builtin pointee

static constexpr int B_ = 512;
static constexpr int T_ = 2048;   // power of two -> cheap div/mod
static constexpr int D_ = 10;
static constexpr int H_ = 20;
static constexpr int BHi = B_ * H_;                  // 10,240
static constexpr int THi = T_ * H_;                  // 40,960
static constexpr long long BH_ = (long long)BHi;

#if __has_builtin(__builtin_amdgcn_global_load_async_to_lds_b128)
#define HAS_ASYNC_LDS 1
typedef __attribute__((address_space(1))) v4i_b* glob_v4i_ptr;
typedef __attribute__((address_space(3))) v4i_b* lds_v4i_ptr;
#endif

__device__ __forceinline__ void wait_async0() {
#if __has_builtin(__builtin_amdgcn_s_wait_asynccnt)
    __builtin_amdgcn_s_wait_asynccnt(0);
#elif defined(HAS_ASYNC_LDS)
    asm volatile("s_wait_asynccnt 0" ::: "memory");
#endif
}

__device__ __forceinline__ float p2scale(float m) {
    // power-of-two scale covering maxabs: exp2(ceil(log2(max(m,1e-10)))) / 128
    return exp2f(ceilf(log2f(fmaxf(m, 1e-10f)))) * (1.0f / 128.0f);
}

__device__ __forceinline__ float wave_max32(float v) {
    #pragma unroll
    for (int m = 16; m > 0; m >>= 1)
        v = fmaxf(v, __shfl_xor(v, m, 32));
    return v;
}

__device__ __forceinline__ float max32_lds(const float* r) {
    float m = r[0];
    #pragma unroll
    for (int i = 1; i < 32; ++i) m = fmaxf(m, r[i]);
    return m;
}

// Stage one 40KB xw slab (10240 f32) into LDS: 2560 x 16B chunks across 1024 threads.
__device__ __forceinline__ void stage_xw_slab(const float* __restrict__ src,
                                              float* dst_lds, int tid) {
#ifdef HAS_ASYNC_LDS
    for (int c = tid; c < 2560; c += 1024) {
        glob_v4i_ptr gp = (glob_v4i_ptr)(unsigned long long)(const void*)(src + 4 * c);
        lds_v4i_ptr  lp = (lds_v4i_ptr)(unsigned int)(unsigned long long)(void*)(dst_lds + 4 * c);
        __builtin_amdgcn_global_load_async_to_lds_b128(gp, lp, 0, 0);
    }
#else
    for (int c = tid; c < 2560; c += 1024) {
        ((float4*)dst_lds)[c] = ((const float4*)src)[c];
    }
#endif
}

// ---------------------------------------------------------------- K0: init
__global__ void init_slots_kernel(unsigned* slots) {
    if (threadIdx.x < 8) slots[threadIdx.x] = 0u;
}

// ---------------------------------------------------------------- K1: absmax
__global__ void absmax_kernel(const float* __restrict__ p, long long n, unsigned* slot) {
    __shared__ float red[256];
    float lm = 0.0f;
    for (long long i = (long long)blockIdx.x * blockDim.x + threadIdx.x; i < n;
         i += (long long)gridDim.x * blockDim.x)
        lm = fmaxf(lm, fabsf(p[i]));
    red[threadIdx.x] = lm;
    __syncthreads();
    #pragma unroll
    for (int s = 128; s > 0; s >>= 1) {
        if ((int)threadIdx.x < s) red[threadIdx.x] = fmaxf(red[threadIdx.x], red[threadIdx.x + s]);
        __syncthreads();
    }
    if (threadIdx.x == 0) atomicMax(slot, __float_as_uint(red[0]));  // nonneg: uint order == float order
}

// ---------------------------------------------------------------- K2: scales + bias quant
__global__ void finalize_kernel(const float* __restrict__ b, const unsigned* __restrict__ slots,
                                float* __restrict__ scales, float* __restrict__ bq) {
    int tid = threadIdx.x;
    if (tid < 8) scales[tid] = p2scale(__uint_as_float(slots[tid]));
    __syncthreads();
    if (tid < 32) {
        float v = 0.0f;
        if (tid < H_) {
            float sb = scales[4];
            float q = rintf(b[tid] / sb);
            q = fminf(fmaxf(q, -128.0f), 127.0f);
            v = q * sb;  // dequantized bias value
        }
        bq[tid] = v;
    }
}

// ---------------------------------------------------------------- K3: xW GEMM via WMMA f16
// 256 threads = 8 waves; each wave: one 16-row M tile x 32 cols (2 WMMA tiles), K 10->32.
// Quantized values exact in f16; f32 accumulation exact. Output stored [T, B, H].
__global__ __launch_bounds__(256) void xw_gemm_kernel(
    const float* __restrict__ x, const float* __restrict__ Wi,
    const float* __restrict__ scales, const float* __restrict__ bq,
    float* __restrict__ xw) {
    __shared__ _Float16 BsT[32][32];       // [N][K]: BsT[n][k] = q(Wi[n][k]) -> contiguous frag reads
    __shared__ _Float16 As[8][16][32];     // per-wave A tile (rows x K), zero-padded

    const int tid  = threadIdx.x;
    const int wave = tid >> 5;
    const int lane = tid & 31;
    const float sx = scales[0];
    const float sw = scales[2];

    for (int i = tid; i < 32 * 32; i += 256) {
        int n = i >> 5, k = i & 31;
        float v = 0.0f;
        if (k < D_ && n < H_) {
            float q = rintf(Wi[n * D_ + k] / sw);
            v = fminf(fmaxf(q, -128.0f), 127.0f);
        }
        BsT[n][k] = (_Float16)v;
    }

    const long long m0 = ((long long)blockIdx.x * 8 + wave) * 16;

    for (int i = lane; i < 16 * 32; i += 32) {
        int r = i >> 5, k = i & 31;
        float v = 0.0f;
        if (k < D_) {
            float q = rintf(x[(m0 + r) * D_ + k] / sx);
            v = fminf(fmaxf(q, -128.0f), 127.0f);
        }
        As[wave][r][k] = (_Float16)v;
    }
    __syncthreads();

    // A fragment (16-bit A 16x32 layout) -> vectorized b128 LDS reads
    v16h A;
    #pragma unroll
    for (int v = 0; v < 8; ++v) {
        int kb = ((lane < 16) ? 0 : 16) + 2 * v;
        A[2 * v]     = As[wave][lane & 15][kb];
        A[2 * v + 1] = As[wave][lane & 15][kb + 1];
    }

    const float ss = sx * sw;
    #pragma unroll
    for (int nt = 0; nt < 2; ++nt) {
        // B fragment: lanes 0-15 K=0..15, lanes 16-31 K=16..31; contiguous in BsT[n][*]
        int n  = nt * 16 + (lane & 15);
        int kb = (lane < 16) ? 0 : 16;
        v16h Bf;
        #pragma unroll
        for (int v = 0; v < 8; ++v) {
            Bf[2 * v]     = BsT[n][kb + 2 * v];
            Bf[2 * v + 1] = BsT[n][kb + 2 * v + 1];
        }
        v8f C = {};
        C = __builtin_amdgcn_wmma_f32_16x16x32_f16(false, A, false, Bf, (short)0, C, false, false);

        if (n < H_) {
            #pragma unroll
            for (int r = 0; r < 8; ++r) {
                long long m  = m0 + r + ((lane < 16) ? 0 : 8);
                long long bb = m >> 11;        // m / T_
                long long t  = m & (T_ - 1);   // m % T_
                xw[t * BH_ + bb * H_ + n] = ss * C[r] + bq[n];
            }
        }
    }
}

// ---------------------------------------------------------------- K4: persistent scan
// One workgroup (1024 threads, 32 waves). h state quantized in LDS; Wh as f16 B-tile in LDS;
// xw slabs double-buffered in LDS via async-to-LDS (prefetched one full step ahead).
// Per step: 2 WMMAs/wave, two tensor-wide maxabs reductions (3 barriers total), quant+tanh+quant.
__global__ __launch_bounds__(1024) void rnn_scan_kernel(
    const float* __restrict__ xw, const float* __restrict__ h0,
    const float* __restrict__ Wh, const float* __restrict__ scales,
    float* __restrict__ out) {
    __shared__ _Float16 qh[512][32];                    // 32 KB quantized h grid values
    __shared__ _Float16 BwT[32][32];                    // 2 KB: BwT[n][k] = q(Wh[n][k])
    __shared__ alignas(16) float xbuf[2][BHi];          // 80 KB double-buffered xw slabs
    __shared__ alignas(16) float red1[32];
    __shared__ alignas(16) float red2[32];

    const int tid  = threadIdx.x;
    const int wave = tid >> 5;
    const int lane = tid & 31;
    const float s_h0 = scales[1];
    const float s_wh = scales[3];

    // kick off slab 0 ASAP
    stage_xw_slab(xw, xbuf[0], tid);

    for (int i = tid; i < 32 * 32; i += 1024) {
        int n = i >> 5, k = i & 31;
        float v = 0.0f;
        if (k < H_ && n < H_) {
            float q = rintf(Wh[n * H_ + k] / s_wh);
            v = fminf(fmaxf(q, -128.0f), 127.0f);
        }
        BwT[n][k] = (_Float16)v;
    }
    for (int i = tid; i < 512 * 32; i += 1024) {
        int r = i >> 5, k = i & 31;
        float v = 0.0f;
        if (k < H_) {
            float q = rintf(h0[r * H_ + k] / s_h0);
            v = fminf(fmaxf(q, -128.0f), 127.0f);
        }
        qh[r][k] = (_Float16)v;
    }
    wait_async0();
    __syncthreads();

    const int mrow = wave * 16;
    float sh = s_h0;

    for (int t = 0; t < T_; ++t) {
        // prefetch next slab a full step ahead (buffer freed by previous step's end barrier)
        if (t + 1 < T_)
            stage_xw_slab(xw + (long long)(t + 1) * BH_, xbuf[(t + 1) & 1], tid);
        const float* xb = xbuf[t & 1];
        const float shw = sh * s_wh;

        // A fragment from quantized h state (b128 LDS reads)
        v16h A;
        #pragma unroll
        for (int v = 0; v < 8; ++v) {
            int kb = ((lane < 16) ? 0 : 16) + 2 * v;
            A[2 * v]     = qh[mrow + (lane & 15)][kb];
            A[2 * v + 1] = qh[mrow + (lane & 15)][kb + 1];
        }

        float vals[2][8];
        float lmax = 0.0f;
        #pragma unroll
        for (int nt = 0; nt < 2; ++nt) {
            int n  = nt * 16 + (lane & 15);
            int kb = (lane < 16) ? 0 : 16;
            v16h Bf;
            #pragma unroll
            for (int v = 0; v < 8; ++v) {
                Bf[2 * v]     = BwT[n][kb + 2 * v];
                Bf[2 * v + 1] = BwT[n][kb + 2 * v + 1];
            }
            v8f C = {};
            C = __builtin_amdgcn_wmma_f32_16x16x32_f16(false, A, false, Bf, (short)0, C, false, false);

            #pragma unroll
            for (int r = 0; r < 8; ++r) {
                int row = mrow + r + ((lane < 16) ? 0 : 8);
                float a = 0.0f;
                if (n < H_) a = xb[row * H_ + n] + shw * C[r];
                vals[nt][r] = a;
                lmax = fmaxf(lmax, fabsf(a));
            }
        }

        // reduction 1: maxabs(acc) -> gate quant scale (uniform across threads)
        if (lane == 0) red1[wave] = wave_max32(lmax);
        __syncthreads();
        const float sa = p2scale(max32_lds(red1));

        // quantize acc (STE forward value), tanh
        float hn[2][8];
        float lmax2 = 0.0f;
        #pragma unroll
        for (int nt = 0; nt < 2; ++nt) {
            int n = nt * 16 + (lane & 15);
            #pragma unroll
            for (int r = 0; r < 8; ++r) {
                float q = rintf(vals[nt][r] / sa);
                q = fminf(fmaxf(q, -128.0f), 127.0f);
                float hv = (n < H_) ? tanhf(q * sa) : 0.0f;
                hn[nt][r] = hv;
                lmax2 = fmaxf(lmax2, fabsf(hv));
            }
        }

        // reduction 2: maxabs(h_new) -> io quant scale
        if (lane == 0) red2[wave] = wave_max32(lmax2);
        __syncthreads();
        const float s2 = p2scale(max32_lds(red2));

        // quantize h_new into state; write output directly in [B,T,H]
        #pragma unroll
        for (int nt = 0; nt < 2; ++nt) {
            int n = nt * 16 + (lane & 15);
            #pragma unroll
            for (int r = 0; r < 8; ++r) {
                int row = mrow + r + ((lane < 16) ? 0 : 8);
                float q = rintf(hn[nt][r] / s2);
                q = fminf(fmaxf(q, -128.0f), 127.0f);
                if (n >= H_) q = 0.0f;
                qh[row][n] = (_Float16)q;
                if (n < H_) out[(long long)row * THi + t * H_ + n] = q * s2;
            }
        }
        sh = s2;

        wait_async0();     // next slab fully in LDS before anyone crosses the barrier
        __syncthreads();   // protects qh + xbuf for next step
    }
}

// ---------------------------------------------------------------- launch
extern "C" void kernel_launch(void* const* d_in, const int* in_sizes, int n_in,
                              void* d_out, int out_size, void* d_ws, size_t ws_size,
                              hipStream_t stream) {
    const float* x  = (const float*)d_in[0];
    const float* h0 = (const float*)d_in[1];
    const float* Wi = (const float*)d_in[2];
    const float* Wh = (const float*)d_in[3];
    const float* b  = (const float*)d_in[4];
    float* out = (float*)d_out;

    char* ws = (char*)d_ws;
    unsigned* slots = (unsigned*)ws;                 // 8 x u32  @ 0
    float* scales   = (float*)(ws + 64);             // 8 x f32
    float* bq       = (float*)(ws + 128);            // 32 x f32
    float* xw       = (float*)(ws + 256);            // T*B*H f32 (80 MB), [T,B,H]

    init_slots_kernel<<<1, 64, 0, stream>>>(slots);

    absmax_kernel<<<2048, 256, 0, stream>>>(x,  (long long)B_ * T_ * D_, slots + 0);
    absmax_kernel<<<64,   256, 0, stream>>>(h0, (long long)B_ * H_,      slots + 1);
    absmax_kernel<<<1,    256, 0, stream>>>(Wi, (long long)H_ * D_,      slots + 2);
    absmax_kernel<<<1,    256, 0, stream>>>(Wh, (long long)H_ * H_,      slots + 3);
    absmax_kernel<<<1,    256, 0, stream>>>(b,  (long long)H_,           slots + 4);

    finalize_kernel<<<1, 64, 0, stream>>>(b, slots, scales, bq);

    // 1,048,576 rows / (8 waves * 16 rows) = 8192 blocks
    xw_gemm_kernel<<<8192, 256, 0, stream>>>(x, Wi, scales, bq, xw);

    rnn_scan_kernel<<<1, 1024, 0, stream>>>(xw, h0, Wh, scales, out);
}
